// PointNetSetAbstraction_11965778886750
// MI455X (gfx1250) — matrix operand order repaired and checked by reading, not compiled, not run on hardware
//
#include <hip/hip_runtime.h>
#include <hip/hip_bf16.h>

typedef _Float16 v8h  __attribute__((ext_vector_type(8)));
typedef _Float16 v16h __attribute__((ext_vector_type(16)));
typedef float    v8f  __attribute__((ext_vector_type(8)));

#define BATCH    16
#define NPTS     8192
#define CIN      64
#define NPOINT   1024
#define NSAMPLE  32
#define RADIUS2  0.04f
#define M_TOTAL  (BATCH * NPOINT * NSAMPLE)   /* 524288 rows */
#define KP0      96                           /* 67 padded to 3x32 */
#define EPSF     1e-5f

// ---------------------------------------------------------------------------
// Kernel 1: farthest point sampling. One block per batch; 8 points per thread
// held entirely in registers (coords + running min-dist). Per iteration:
// broadcast current centroid from its owner thread via LDS, update distances,
// 2-level argmax (wave32 shuffle reduce, then LDS across 32 waves).
// ---------------------------------------------------------------------------
__global__ __launch_bounds__(1024)
void fps_kernel(const float* __restrict__ xyz,
                float* __restrict__ newxyz,     // ws copy (B,NPOINT,3)
                float* __restrict__ out_xyz)    // d_out   (B,NPOINT,3)
{
    const int b = blockIdx.x;
    const int t = threadIdx.x;

    __shared__ float s_val[32];
    __shared__ int   s_idx[32];
    __shared__ float s_c[3];
    __shared__ int   s_win;

    float px[8], py[8], pz[8], dist[8];
#pragma unroll
    for (int k = 0; k < 8; ++k) {
        int i = t + k * 1024;
        const float* p = xyz + ((size_t)b * NPTS + i) * 3;
        px[k] = p[0]; py[k] = p[1]; pz[k] = p[2];
        dist[k] = 1e10f;
    }

    int cur = 0;
    for (int it = 0; it < NPOINT; ++it) {
        // broadcast centroid coords from the owning thread (registers -> LDS)
        if (t == (cur & 1023)) {
            int k = cur >> 10;
            s_c[0] = px[k]; s_c[1] = py[k]; s_c[2] = pz[k];
        }
        __syncthreads();
        float cx = s_c[0], cy = s_c[1], cz = s_c[2];
        if (t == 0) {
            float* o  = out_xyz + ((size_t)b * NPOINT + it) * 3;
            float* o2 = newxyz  + ((size_t)b * NPOINT + it) * 3;
            o[0]  = cx; o[1]  = cy; o[2]  = cz;
            o2[0] = cx; o2[1] = cy; o2[2] = cz;
        }

        float bv = -1.0f; int bi = 0x7fffffff;
#pragma unroll
        for (int k = 0; k < 8; ++k) {
            float dx = px[k] - cx, dy = py[k] - cy, dz = pz[k] - cz;
            float d  = dx * dx + dy * dy + dz * dz;
            float nd = fminf(dist[k], d);
            dist[k] = nd;
            int i = t + k * 1024;
            if (nd > bv || (nd == bv && i < bi)) { bv = nd; bi = i; }
        }
        // wave32 argmax (tie-break: lowest index, matches jnp.argmax)
#pragma unroll
        for (int off = 16; off > 0; off >>= 1) {
            float ov = __shfl_down(bv, off);
            int   oi = __shfl_down(bi, off);
            if (ov > bv || (ov == bv && oi < bi)) { bv = ov; bi = oi; }
        }
        if ((t & 31) == 0) { s_val[t >> 5] = bv; s_idx[t >> 5] = bi; }
        __syncthreads();
        if (t < 32) {
            float v = s_val[t]; int i = s_idx[t];
#pragma unroll
            for (int off = 16; off > 0; off >>= 1) {
                float ov = __shfl_down(v, off);
                int   oi = __shfl_down(i, off);
                if (ov > v || (ov == v && oi < i)) { v = ov; i = oi; }
            }
            if (t == 0) s_win = i;
        }
        __syncthreads();
        cur = s_win;
    }
}

// ---------------------------------------------------------------------------
// Kernel 2: ball query (ordered compaction via ballot) + gather + concat into
// f16 staging rows [feat(64) | relxyz(3) | zero-pad -> KP0]. One wave per
// centroid; lane k owns sample k.
// ---------------------------------------------------------------------------
__global__ __launch_bounds__(256)
void ball_gather(const float* __restrict__ xyz, const float* __restrict__ points,
                 const float* __restrict__ newxyz, _Float16* __restrict__ x0)
{
    __shared__ int s_slots[8][NSAMPLE];
    const int lane = threadIdx.x & 31;
    const int wv   = threadIdx.x >> 5;
    const long cent = (long)blockIdx.x * 8 + wv;      // 0 .. B*NPOINT-1
    const int  b    = (int)(cent >> 10);

    const float* cc = newxyz + (size_t)cent * 3;
    const float cx = cc[0], cy = cc[1], cz = cc[2];
    const float* xb = xyz + (size_t)b * NPTS * 3;

    int count = 0;
    for (int base = 0; base < NPTS && count < NSAMPLE; base += 32) {
        int i = base + lane;
        const float* p = xb + (size_t)i * 3;
        float dx = p[0] - cx, dy = p[1] - cy, dz = p[2] - cz;
        bool inR = (dx * dx + dy * dy + dz * dz) <= RADIUS2;
        unsigned m   = (unsigned)__ballot(inR);
        int      pos = count + __popc(m & ((1u << lane) - 1u));
        if (inR && pos < NSAMPLE) s_slots[wv][pos] = i;
        count += __popc(m);
    }
    if (count > NSAMPLE) count = NSAMPLE;
    // LDS ops are in-order within a wave; pad missing samples with first idx
    int idx = s_slots[wv][lane < count ? lane : 0];

    const float* pt = xb + (size_t)idx * 3;
    float rx = pt[0] - cx, ry = pt[1] - cy, rz = pt[2] - cz;

    _Float16* row = x0 + ((size_t)cent * NSAMPLE + lane) * KP0;
    const float* fb = points + (size_t)b * CIN * NPTS + idx;
#pragma unroll 8
    for (int c = 0; c < CIN; ++c) row[c] = (_Float16)fb[(size_t)c * NPTS];
    row[64] = (_Float16)rx; row[65] = (_Float16)ry; row[66] = (_Float16)rz;
#pragma unroll
    for (int c = 67; c < KP0; ++c) row[c] = (_Float16)0.0f;
}

// ---------------------------------------------------------------------------
// Small helpers: pack f32 weights (o, cin) -> f16 (o, KP) zero-padded rows,
// and zero the per-layer stat accumulators (3 layers x 256 floats).
// ---------------------------------------------------------------------------
__global__ void pack_w(const float* __restrict__ w, _Float16* __restrict__ wf,
                       int nout, int cin, int kp)
{
    int i = blockIdx.x * 256 + threadIdx.x;
    if (i >= nout * kp) return;
    int o = i / kp, k = i % kp;
    wf[i] = (_Float16)(k < cin ? w[o * cin + k] : 0.0f);
}

__global__ void zero_stats(float* __restrict__ s)
{
    int i = threadIdx.x;
    if (i < 768) s[i] = 0.0f;
}

// ---------------------------------------------------------------------------
// Kernel 3: WMMA GEMM  Y[M,NOUT](f16) = X[M,KP](f16) @ W[NOUT,KP]^T, f32 acc,
// fused per-channel sum / sum-of-squares via LDS atomics + one global flush.
// One wave per 16x16 output tile. Fragment loads use contiguous k order (same for A and B):
//   A: lane l (m = l&15, khalf = l>>4) holds K = khalf*16+{0..7} and
//      khalf*16+8+{0..7}  -> two contiguous 16B loads per 32-chunk.
//   B: lane l (n = l&15) holds K = (l>>4)*16+{0..15} -> one 32B load.
// ---------------------------------------------------------------------------
template <int KP, int NOUT>
__global__ __launch_bounds__(256)
void gemm_stats(const _Float16* __restrict__ X, const _Float16* __restrict__ W,
                _Float16* __restrict__ Y, float* __restrict__ stats /* [2*NOUT] */)
{
    __shared__ float s_sum[NOUT];
    __shared__ float s_sq[NOUT];
    for (int i = threadIdx.x; i < NOUT; i += 256) { s_sum[i] = 0.0f; s_sq[i] = 0.0f; }
    __syncthreads();

    const int lane = threadIdx.x & 31;
    const int wv   = threadIdx.x >> 5;
    const int l15  = lane & 15;
    const int lh   = lane >> 4;
    constexpr int TN = NOUT / 16;

    long wid = (long)blockIdx.x * 8 + wv;
    int  tn  = (int)(wid % TN);
    long tm  = wid / TN;

    const _Float16* xr = X + ((size_t)(tm * 16 + l15)) * KP + lh * 16;
    const _Float16* wr = W + ((size_t)(tn * 16 + l15)) * KP + lh * 16;

    v8f acc = {};
#pragma unroll
    for (int kc = 0; kc < KP / 32; ++kc) {
        v8h lo = *(const v8h*)(xr + kc * 32);
        v8h hi = *(const v8h*)(xr + kc * 32 + 8);
        v16h a = __builtin_shufflevector(lo, hi, 0, 1, 2, 3, 4, 5, 6, 7,
                                                 8, 9, 10, 11, 12, 13, 14, 15);
        v16h bm = *(const v16h*)(wr + kc * 32);
        acc = __builtin_amdgcn_wmma_f32_16x16x32_f16(false, a, false, bm,
                                                     (short)0, acc, false, false);
    }

    const int    n  = tn * 16 + l15;
    const size_t mb = (size_t)tm * 16 + lh * 8;   // D layout: VGPR r -> m = r (+8 for lanes>=16)
    float sv = 0.0f, sq = 0.0f;
#pragma unroll
    for (int r = 0; r < 8; ++r) {
        float v = acc[r];
        sv += v; sq += v * v;
        Y[(mb + r) * NOUT + n] = (_Float16)v;
    }
    atomicAdd(&s_sum[n], sv);   // ds_add_f32
    atomicAdd(&s_sq[n],  sq);
    __syncthreads();
    for (int i = threadIdx.x; i < NOUT; i += 256) {
        atomicAdd(&stats[i],        s_sum[i]);   // global_atomic_add_f32
        atomicAdd(&stats[NOUT + i], s_sq[i]);
    }
}

// ---------------------------------------------------------------------------
// Kernel 4: batch-norm (batch statistics) + ReLU, f16 -> f16 (C = 64).
// mean = S/M, var = SQ/M - mean^2; bias provably cancels, g/beta applied.
// ---------------------------------------------------------------------------
__global__ __launch_bounds__(256)
void norm_relu64(const _Float16* __restrict__ Y, _Float16* __restrict__ X,
                 const float* __restrict__ stats, const float* __restrict__ g,
                 const float* __restrict__ beta)
{
    size_t i = (size_t)blockIdx.x * 256 + threadIdx.x;   // over M*64
    int c = (int)(i & 63);
    const float invM = 1.0f / (float)M_TOTAL;
    float mean = stats[c] * invM;
    float var  = stats[64 + c] * invM - mean * mean;
    float sc   = g[c] * rsqrtf(var + EPSF);
    float sh   = beta[c] - mean * sc;
    float v = (float)Y[i] * sc + sh;
    X[i] = (_Float16)fmaxf(v, 0.0f);
}

// ---------------------------------------------------------------------------
// Kernel 5: layer-3 norm + ReLU fused with max over the 32 samples, writing
// the transposed (B,128,NPOINT) f32 output. relu(max) == max(relu) clamp.
// ---------------------------------------------------------------------------
__global__ __launch_bounds__(256)
void norm_max(const _Float16* __restrict__ Y, float* __restrict__ out,
              const float* __restrict__ stats, const float* __restrict__ g,
              const float* __restrict__ beta)
{
    size_t id = (size_t)blockIdx.x * 256 + threadIdx.x;  // B*NPOINT*128
    int    c  = (int)(id & 127);
    size_t bs = id >> 7;
    const float invM = 1.0f / (float)M_TOTAL;
    float mean = stats[c] * invM;
    float var  = stats[128 + c] * invM - mean * mean;
    float sc   = g[c] * rsqrtf(var + EPSF);
    float sh   = beta[c] - mean * sc;
    float mx = -1e30f;
#pragma unroll 8
    for (int k = 0; k < NSAMPLE; ++k) {
        float v = (float)Y[(bs * NSAMPLE + k) * 128 + c] * sc + sh;
        mx = fmaxf(mx, v);
    }
    mx = fmaxf(mx, 0.0f);
    int b = (int)(bs >> 10), s = (int)(bs & 1023);
    out[((size_t)b * 128 + c) * NPOINT + s] = mx;
}

// ---------------------------------------------------------------------------
extern "C" void kernel_launch(void* const* d_in, const int* in_sizes, int n_in,
                              void* d_out, int out_size, void* d_ws, size_t ws_size,
                              hipStream_t stream)
{
    (void)in_sizes; (void)n_in; (void)out_size; (void)ws_size;
    const float* xyz    = (const float*)d_in[0];
    const float* points = (const float*)d_in[1];
    const float* w0  = (const float*)d_in[2];
    const float* g0  = (const float*)d_in[4];
    const float* be0 = (const float*)d_in[5];
    const float* w1  = (const float*)d_in[6];
    const float* g1  = (const float*)d_in[8];
    const float* be1 = (const float*)d_in[9];
    const float* w2  = (const float*)d_in[10];
    const float* g2  = (const float*)d_in[12];
    const float* be2 = (const float*)d_in[13];

    float* out_xyz = (float*)d_out;                       // (B,NPOINT,3)
    float* out_pts = out_xyz + (size_t)BATCH * NPOINT * 3; // (B,128,NPOINT)

    char* ws = (char*)d_ws;
    size_t off = 0;
    auto carve = [&](size_t bytes) -> char* {
        char* p = ws + off;
        off += (bytes + 255) & ~(size_t)255;
        return p;
    };
    float*    newxyz = (float*)carve((size_t)BATCH * NPOINT * 3 * sizeof(float));
    float*    stats  = (float*)carve(768 * sizeof(float));             // 3 x 256
    _Float16* wf0    = (_Float16*)carve((size_t)64  * KP0 * sizeof(_Float16));
    _Float16* wf1    = (_Float16*)carve((size_t)64  * 64  * sizeof(_Float16));
    _Float16* wf2    = (_Float16*)carve((size_t)128 * 64  * sizeof(_Float16));
    _Float16* x0     = (_Float16*)carve((size_t)M_TOTAL * KP0 * sizeof(_Float16));
    _Float16* ybuf   = (_Float16*)carve((size_t)M_TOTAL * 128 * sizeof(_Float16));
    _Float16* x12    = (_Float16*)carve((size_t)M_TOTAL * 64  * sizeof(_Float16));

    fps_kernel<<<BATCH, 1024, 0, stream>>>(xyz, newxyz, out_xyz);

    pack_w<<<(64 * KP0 + 255) / 256, 256, 0, stream>>>(w0, wf0, 64, 67, KP0);
    pack_w<<<(64 * 64  + 255) / 256, 256, 0, stream>>>(w1, wf1, 64, 64, 64);
    pack_w<<<(128 * 64 + 255) / 256, 256, 0, stream>>>(w2, wf2, 128, 64, 64);
    zero_stats<<<1, 768, 0, stream>>>(stats);

    ball_gather<<<(BATCH * NPOINT) / 8, 256, 0, stream>>>(xyz, points, newxyz, x0);

    // Layer 1: (M x 96) @ (96 x 64)
    gemm_stats<KP0, 64><<<(M_TOTAL / 16) * 4 / 8, 256, 0, stream>>>(x0, wf0, ybuf, stats);
    norm_relu64<<<(int)((size_t)M_TOTAL * 64 / 256), 256, 0, stream>>>(ybuf, x12, stats, g0, be0);

    // Layer 2: (M x 64) @ (64 x 64)
    gemm_stats<64, 64><<<(M_TOTAL / 16) * 4 / 8, 256, 0, stream>>>(x12, wf1, ybuf, stats + 256);
    norm_relu64<<<(int)((size_t)M_TOTAL * 64 / 256), 256, 0, stream>>>(ybuf, x12, stats + 256, g1, be1);

    // Layer 3: (M x 64) @ (64 x 128), then norm + maxpool + transpose
    gemm_stats<64, 128><<<(M_TOTAL / 16) * 8 / 8, 256, 0, stream>>>(x12, wf2, ybuf, stats + 512);
    norm_max<<<(BATCH * NPOINT * 128) / 256, 256, 0, stream>>>(ybuf, out_pts, stats + 512, g2, be2);
}
